// SceneRealitivePose_63393717289599
// MI455X (gfx1250) — matrix operand order repaired and checked by reading, not compiled, not run on hardware
//
#include <hip/hip_runtime.h>
#include <hip/hip_bf16.h>

typedef _Float16 v16h __attribute__((ext_vector_type(16)));
typedef _Float16 v8h  __attribute__((ext_vector_type(8)));
typedef float    v8f  __attribute__((ext_vector_type(8)));
typedef unsigned int v4u __attribute__((ext_vector_type(4)));
typedef int      v4i  __attribute__((ext_vector_type(4)));
typedef int      v8i  __attribute__((ext_vector_type(8)));

#define N_AGENT 1024
#define N_MAP   3072
#define N_ALL   4096
#define D       256
#define NKNN    20
#define DFF     2048

#if defined(__has_builtin)
#  if __has_builtin(__builtin_amdgcn_tensor_load_to_lds) && \
      __has_builtin(__builtin_amdgcn_s_wait_tensorcnt)
#    define USE_TDM 1
#  endif
#endif
#ifndef USE_TDM
#  define USE_TDM 0
#endif

// ---------------------------------------------------------------------------
// 1) Top-20 smallest of scene[2, a, 1024:4096] per agent row a.
//    (DIAG_FILL hits cols < 1024 for these rows, so it never lands in the
//    slice.)  Tie-break: lower index first, matching lax.top_k on -rd.
// ---------------------------------------------------------------------------
__global__ __launch_bounds__(256)
void topk_kernel(const float* __restrict__ scene, int* __restrict__ idx_out)
{
    __shared__ float vals[N_MAP];
    __shared__ float rv[256];
    __shared__ int   ri[256];
    const int a = blockIdx.x;
    const int t = threadIdx.x;
    const float* row = scene + (size_t)2 * N_ALL * N_ALL + (size_t)a * N_ALL + N_AGENT;
    for (int j = t; j < N_MAP; j += 256) vals[j] = row[j];
    __syncthreads();
    for (int it = 0; it < NKNN; ++it) {
        float bv = 3.4e38f; int bi = 0x7fffffff;
        for (int j = t; j < N_MAP; j += 256) {
            float v = vals[j];
            if (v < bv || (v == bv && j < bi)) { bv = v; bi = j; }
        }
        rv[t] = bv; ri[t] = bi;
        __syncthreads();
        for (int s = 128; s > 0; s >>= 1) {
            if (t < s) {
                float v2 = rv[t + s]; int i2 = ri[t + s];
                if (v2 < rv[t] || (v2 == rv[t] && i2 < ri[t])) { rv[t] = v2; ri[t] = i2; }
            }
            __syncthreads();
        }
        const int w = ri[0];
        if (t == 0) { idx_out[a * NKNN + it] = w; vals[w] = 3.4e38f; }
        __syncthreads();
    }
}

// ---------------------------------------------------------------------------
// 2) Gather rel_pose at knn indices and compute 256-dim pose embedding (f16).
// ---------------------------------------------------------------------------
__global__ __launch_bounds__(256)
void rpe_kernel(const float* __restrict__ rel_pose, const int* __restrict__ idx,
                _Float16* __restrict__ rpe)
{
    const int a = blockIdx.x;
    const int j = blockIdx.y;
    const int t = threadIdx.x;
    const int col = idx[a * NKNN + j];
    const size_t base = ((size_t)a * N_ALL + N_AGENT + col) * 3;
    const float x = rel_pose[base], y = rel_pose[base + 1], yaw = rel_pose[base + 2];
    float vq[4];
    vq[0] = x; vq[1] = y; vq[2] = cosf(yaw); vq[3] = sinf(yaw);
    const int qd = t >> 6;
    const int u  = t & 63;
    const int fi = u & 31;
    const int isSin = u >> 5;
    const float theta = (qd < 2) ? 1000.0f : 10.0f;
    const float f = powf(theta, -(float)fi / 32.0f);
    const float pe = vq[qd] * f;
    const float o = isSin ? sinf(pe) : cosf(pe);
    rpe[((size_t)(a * NKNN + j)) * D + t] = (_Float16)o;
}

// ---------------------------------------------------------------------------
// 3a) f32 -> f16 flat convert
// ---------------------------------------------------------------------------
__global__ __launch_bounds__(256)
void conv_f16_kernel(const float* __restrict__ in, _Float16* __restrict__ out, int n)
{
    int i = blockIdx.x * 256 + threadIdx.x;
    if (i < n) out[i] = (_Float16)in[i];
}

// 3b) f32 (K x N) -> f16 transposed (N x K)
__global__ __launch_bounds__(256)
void conv_transpose_kernel(const float* __restrict__ W, _Float16* __restrict__ Wt,
                           int K, int N)
{
    int i = blockIdx.x * 256 + threadIdx.x;
    if (i >= K * N) return;
    int k = i / N, n = i % N;
    Wt[(size_t)n * K + k] = (_Float16)W[i];
}

// ---------------------------------------------------------------------------
// 4) WMMA GEMM: C[M,N] = A[M,K](f16,row) @ Bt[N,K](f16,row) (+bias)(relu).
//    Block = 8 waves; block tile = 128(M) x 64(N); each wave owns 16x64.
//    Bt slab (64 rows x 256 k) is staged in LDS per 256-k chunk via the
//    Tensor Data Mover (tensor_load_to_lds + s_wait_tensorcnt) when the
//    builtin is available, else via manual load->ds_store staging.
//    A fragments come straight from global (b128 pairs) and are reused
//    across the 4 N-tiles.
// ---------------------------------------------------------------------------
__global__ __launch_bounds__(256)
void gemm16_wmma(const _Float16* __restrict__ A, const _Float16* __restrict__ Bt,
                 const float* __restrict__ bias, float* __restrict__ C32,
                 _Float16* __restrict__ C16, int M, int N, int K, int relu)
{
    __shared__ _Float16 bs[64 * 256];          // 32 KB slab: 64 n-rows x 256 k
    const int wave = threadIdx.x >> 5;
    const int lane = threadIdx.x & 31;
    const int nBase = blockIdx.x * 64;
    const int mBase = (blockIdx.y * 8 + wave) * 16;
    const int hi    = lane >> 4;               // 0 or 1 (k-half select)
    const _Float16* aRow = A + (size_t)(mBase + (lane & 15)) * K + hi * 8;

    v8f acc[4];
    #pragma unroll
    for (int nt = 0; nt < 4; ++nt) acc[nt] = (v8f){};

    for (int kc = 0; kc < K; kc += 256) {
#if USE_TDM
        if (wave == 0) {
            if (lane == 0) {
                const unsigned long long ga =
                    (unsigned long long)(size_t)(const void*)(Bt + (size_t)nBase * K + kc);
                const unsigned ldsa = (unsigned)(size_t)(void*)bs;
                v4u g0;
                g0.x = 1u;                                        // count=1, user D#
                g0.y = ldsa;                                      // lds_addr
                g0.z = (unsigned)(ga & 0xffffffffu);              // global_addr[31:0]
                g0.w = (unsigned)((ga >> 32) & 0x1ffffffu)        // global_addr[56:32]
                     | (2u << 30);                                // type = 2 (image)
                const unsigned t0 = (unsigned)(K >> 1);           // tensor_dim0 (dwords)
                const unsigned t1 = (unsigned)N;                  // tensor_dim1 (rows)
                const unsigned long long s0 = (unsigned long long)(K >> 1); // row stride
                v8i g1;
                g1[0] = (int)(2u << 16);                          // wg_mask=0, data_size=4B
                g1[1] = (int)((t0 & 0xffffu) << 16);              // tensor_dim0[15:0]
                g1[2] = (int)((t0 >> 16) | ((t1 & 0xffffu) << 16));
                g1[3] = (int)((t1 >> 16) | (128u << 16));         // tile_dim0 = 128 dwords
                g1[4] = (int)64u;                                 // tile_dim1 = 64, dim2 = 0
                g1[5] = (int)(unsigned)(s0 & 0xffffffffu);        // stride0[31:0]
                g1[6] = (int)(unsigned)((s0 >> 32) & 0xffffu);    // stride0[47:32]
                g1[7] = 0;
                v4i z4; z4[0] = 0; z4[1] = 0; z4[2] = 0; z4[3] = 0;
#if __clang_major__ >= 23
                v8i z8; 
                #pragma unroll
                for (int zz = 0; zz < 8; ++zz) z8[zz] = 0;
                __builtin_amdgcn_tensor_load_to_lds(g0, g1, z4, z4, z8, 0);
#else
                __builtin_amdgcn_tensor_load_to_lds(g0, g1, z4, z4, 0);
#endif
            }
            __builtin_amdgcn_s_wait_tensorcnt(0);
        }
        __syncthreads();
#else
        // fallback staging: 64 rows x 256 halves, 16B segments, coalesced
        for (int i = threadIdx.x; i < 64 * 32; i += 256) {
            const int n = i >> 5, seg = i & 31;
            *(v8h*)&bs[n * 256 + seg * 8] =
                *(const v8h*)&Bt[(size_t)(nBase + n) * K + kc + seg * 8];
        }
        __syncthreads();
#endif
        #pragma unroll 4
        for (int kk = 0; kk < 256; kk += 32) {
            __builtin_prefetch(aRow + kc + kk + 256, 0, 1);   // global_prefetch_b8
            const v8h a0 = *(const v8h*)(aRow + kc + kk);
            const v8h a1 = *(const v8h*)(aRow + kc + kk + 16);
            v16h a;
            #pragma unroll
            for (int i = 0; i < 8; ++i) { a[i] = a0[i]; a[i + 8] = a1[i]; }
            const int bk = kk + hi * 8;
            #pragma unroll
            for (int nt = 0; nt < 4; ++nt) {
                const _Float16* bp = &bs[(size_t)(nt * 16 + (lane & 15)) * 256 + bk];
                const v8h b0 = *(const v8h*)bp;          // ds_load_b128
                const v8h b1 = *(const v8h*)(bp + 16);   // ds_load_b128
                v16h b;
                #pragma unroll
                for (int i = 0; i < 8; ++i) { b[i] = b0[i]; b[i + 8] = b1[i]; }
                acc[nt] = __builtin_amdgcn_wmma_f32_16x16x32_f16(
                              false, a, false, b, (short)0, acc[nt], false, false);
            }
        }
        __syncthreads();   // slab reuse fence before next chunk
    }

    const int colBase  = nBase + (lane & 15);
    const int rowStart = mBase + hi * 8;
    #pragma unroll
    for (int nt = 0; nt < 4; ++nt) {
        const int col = colBase + nt * 16;
        const float bvv = bias ? bias[col] : 0.0f;
        #pragma unroll
        for (int v = 0; v < 8; ++v) {
            float val = acc[nt][v] + bvv;
            if (relu) val = fmaxf(val, 0.0f);
            const size_t off = (size_t)(rowStart + v) * N + col;
            if (C32) C32[off] = val;
            if (C16) C16[off] = (_Float16)val;
        }
    }
}

// ---------------------------------------------------------------------------
// 5) KNN attention: one wave = one (agent, head); dh == warpSize == 32.
// ---------------------------------------------------------------------------
__global__ __launch_bounds__(256)
void attn_kernel(const float* __restrict__ q, const float* __restrict__ kb,
                 const float* __restrict__ vb, const _Float16* __restrict__ rk,
                 const _Float16* __restrict__ rv, float* __restrict__ o32,
                 _Float16* __restrict__ o16)
{
    const int a    = blockIdx.x;
    const int h    = threadIdx.x >> 5;
    const int lane = threadIdx.x & 31;
    const int d    = h * 32 + lane;
    const size_t rowOff = (size_t)a * D + d;
    const float qv  = q[rowOff] * 0.17677669529663687f;   // 1/sqrt(32)
    const float kbv = kb[rowOff];
    float sc[NKNN];
    #pragma unroll
    for (int j = 0; j < NKNN; ++j) {
        float kk = kbv + (float)rk[((size_t)(a * NKNN + j)) * D + d];
        float p = qv * kk;
        #pragma unroll
        for (int off = 16; off > 0; off >>= 1) p += __shfl_xor(p, off, 32);
        sc[j] = p;
    }
    float mx = sc[0];
    #pragma unroll
    for (int j = 1; j < NKNN; ++j) mx = fmaxf(mx, sc[j]);
    float sum = 0.0f;
    #pragma unroll
    for (int j = 0; j < NKNN; ++j) { sc[j] = expf(sc[j] - mx); sum += sc[j]; }
    const float inv = 1.0f / sum;
    const float vbase = vb[rowOff];
    float out = 0.0f;
    #pragma unroll
    for (int j = 0; j < NKNN; ++j) {
        float vv = vbase + (float)rv[((size_t)(a * NKNN + j)) * D + d];
        out += sc[j] * inv * vv;
    }
    o32[rowOff] = out;
    o16[rowOff] = (_Float16)out;
}

// ---------------------------------------------------------------------------
// 6) out = LayerNorm(X + Y) * g + b ; optional f32 / f16 outputs.
// ---------------------------------------------------------------------------
__global__ __launch_bounds__(256)
void add_ln_kernel(const float* __restrict__ X, const float* __restrict__ Y,
                   const float* __restrict__ g, const float* __restrict__ b,
                   float* __restrict__ out32, _Float16* __restrict__ out16)
{
    __shared__ float red[256];
    const int row = blockIdx.x, t = threadIdx.x;
    const size_t off = (size_t)row * D + t;
    const float v = X[off] + Y[off];
    red[t] = v; __syncthreads();
    for (int s = 128; s > 0; s >>= 1) { if (t < s) red[t] += red[t + s]; __syncthreads(); }
    const float mu = red[0] / (float)D;
    __syncthreads();
    const float dv = v - mu;
    red[t] = dv * dv; __syncthreads();
    for (int s = 128; s > 0; s >>= 1) { if (t < s) red[t] += red[t + s]; __syncthreads(); }
    const float var = red[0] / (float)D;
    const float o = dv * rsqrtf(var + 1e-5f) * g[t] + b[t];
    if (out32) out32[off] = o;
    if (out16) out16[off] = (_Float16)o;
}

// ---------------------------------------------------------------------------
extern "C" void kernel_launch(void* const* d_in, const int* in_sizes, int n_in,
                              void* d_out, int out_size, void* d_ws, size_t ws_size,
                              hipStream_t stream)
{
    (void)in_sizes; (void)n_in; (void)out_size; (void)ws_size;
    const float* actors   = (const float*)d_in[0];
    const float* lanes    = (const float*)d_in[1];
    const float* scene    = (const float*)d_in[2];
    const float* rel_pose = (const float*)d_in[3];
    const float* Wq  = (const float*)d_in[4];   const float* bq  = (const float*)d_in[5];
    const float* Wk  = (const float*)d_in[6];   const float* bk  = (const float*)d_in[7];
    const float* Wv  = (const float*)d_in[8];   const float* bv  = (const float*)d_in[9];
    const float* Wrk = (const float*)d_in[10];  const float* Wrv = (const float*)d_in[11];
    const float* Wo  = (const float*)d_in[12];  const float* bo  = (const float*)d_in[13];
    const float* ln1g = (const float*)d_in[14]; const float* ln1b = (const float*)d_in[15];
    const float* W1  = (const float*)d_in[16];  const float* b1  = (const float*)d_in[17];
    const float* W2  = (const float*)d_in[18];  const float* b2  = (const float*)d_in[19];
    const float* ln2g = (const float*)d_in[20]; const float* ln2b = (const float*)d_in[21];

    char* ws = (char*)d_ws;
    size_t cur = 0;
    auto alloc = [&](size_t bytes) {
        char* p = ws + cur;
        cur += (bytes + 255) & ~(size_t)255;
        return p;
    };
    int*       idx     = (int*)      alloc(N_AGENT * NKNN * 4);
    _Float16*  actorsH = (_Float16*) alloc((size_t)N_AGENT * D * 2);
    _Float16*  WqT  = (_Float16*) alloc((size_t)D * D * 2);
    _Float16*  WkT  = (_Float16*) alloc((size_t)D * D * 2);
    _Float16*  WvT  = (_Float16*) alloc((size_t)D * D * 2);
    _Float16*  WrkT = (_Float16*) alloc((size_t)D * D * 2);
    _Float16*  WrvT = (_Float16*) alloc((size_t)D * D * 2);
    _Float16*  WoT  = (_Float16*) alloc((size_t)D * D * 2);
    _Float16*  W1T  = (_Float16*) alloc((size_t)D * DFF * 2);
    _Float16*  W2T  = (_Float16*) alloc((size_t)D * DFF * 2);
    _Float16*  rpeH = (_Float16*) alloc((size_t)N_AGENT * NKNN * D * 2);
    float*     q32  = (float*)    alloc((size_t)N_AGENT * D * 4);
    float*     kb32 = (float*)    alloc((size_t)N_AGENT * D * 4);
    float*     vb32 = (float*)    alloc((size_t)N_AGENT * D * 4);
    _Float16*  rkH  = (_Float16*) alloc((size_t)N_AGENT * NKNN * D * 2);
    _Float16*  rvH  = (_Float16*) alloc((size_t)N_AGENT * NKNN * D * 2);
    float*     at32 = (float*)    alloc((size_t)N_AGENT * D * 4);
    _Float16*  atH  = (_Float16*) alloc((size_t)N_AGENT * D * 2);
    float*     pr32 = (float*)    alloc((size_t)N_AGENT * D * 4);
    float*     x132 = (float*)    alloc((size_t)N_AGENT * D * 4);
    _Float16*  x1H  = (_Float16*) alloc((size_t)N_AGENT * D * 2);
    _Float16*  hH   = (_Float16*) alloc((size_t)N_AGENT * DFF * 2);
    float*     ff32 = (float*)    alloc((size_t)N_AGENT * D * 4);

    topk_kernel<<<N_AGENT, 256, 0, stream>>>(scene, idx);
    rpe_kernel<<<dim3(N_AGENT, NKNN), 256, 0, stream>>>(rel_pose, idx, rpeH);

    conv_f16_kernel<<<(N_AGENT * D + 255) / 256, 256, 0, stream>>>(actors, actorsH, N_AGENT * D);
    conv_transpose_kernel<<<(D * D + 255) / 256, 256, 0, stream>>>(Wq,  WqT,  D, D);
    conv_transpose_kernel<<<(D * D + 255) / 256, 256, 0, stream>>>(Wk,  WkT,  D, D);
    conv_transpose_kernel<<<(D * D + 255) / 256, 256, 0, stream>>>(Wv,  WvT,  D, D);
    conv_transpose_kernel<<<(D * D + 255) / 256, 256, 0, stream>>>(Wrk, WrkT, D, D);
    conv_transpose_kernel<<<(D * D + 255) / 256, 256, 0, stream>>>(Wrv, WrvT, D, D);
    conv_transpose_kernel<<<(D * D + 255) / 256, 256, 0, stream>>>(Wo,  WoT,  D, D);
    conv_transpose_kernel<<<(D * DFF + 255) / 256, 256, 0, stream>>>(W1, W1T, D, DFF);
    conv_transpose_kernel<<<(D * DFF + 255) / 256, 256, 0, stream>>>(W2, W2T, DFF, D);

    // q/k/v base projections (tgt is broadcast actors, so tgt@Wk == actors@Wk)
    gemm16_wmma<<<dim3(D / 64, N_AGENT / 128), 256, 0, stream>>>(
        actorsH, WqT, bq, q32, nullptr, N_AGENT, D, D, 0);
    gemm16_wmma<<<dim3(D / 64, N_AGENT / 128), 256, 0, stream>>>(
        actorsH, WkT, bk, kb32, nullptr, N_AGENT, D, D, 0);
    gemm16_wmma<<<dim3(D / 64, N_AGENT / 128), 256, 0, stream>>>(
        actorsH, WvT, bv, vb32, nullptr, N_AGENT, D, D, 0);
    // rpe projections: (20480 x 256) @ (256 x 256)
    gemm16_wmma<<<dim3(D / 64, (N_AGENT * NKNN) / 128), 256, 0, stream>>>(
        rpeH, WrkT, nullptr, nullptr, rkH, N_AGENT * NKNN, D, D, 0);
    gemm16_wmma<<<dim3(D / 64, (N_AGENT * NKNN) / 128), 256, 0, stream>>>(
        rpeH, WrvT, nullptr, nullptr, rvH, N_AGENT * NKNN, D, D, 0);

    attn_kernel<<<N_AGENT, 256, 0, stream>>>(q32, kb32, vb32, rkH, rvH, at32, atH);

    gemm16_wmma<<<dim3(D / 64, N_AGENT / 128), 256, 0, stream>>>(
        atH, WoT, bo, pr32, nullptr, N_AGENT, D, D, 0);

    add_ln_kernel<<<N_AGENT, 256, 0, stream>>>(actors, pr32, ln1g, ln1b, x132, x1H);

    gemm16_wmma<<<dim3(DFF / 64, N_AGENT / 128), 256, 0, stream>>>(
        x1H, W1T, b1, nullptr, hH, N_AGENT, DFF, D, 1);
    gemm16_wmma<<<dim3(D / 64, N_AGENT / 128), 256, 0, stream>>>(
        hH, W2T, b2, ff32, nullptr, N_AGENT, D, DFF, 0);

    add_ln_kernel<<<N_AGENT, 256, 0, stream>>>(x132, ff32, ln2g, ln2b, (float*)d_out, nullptr);

    hipMemcpyAsync((float*)d_out + (size_t)N_AGENT * D, lanes,
                   (size_t)N_MAP * D * sizeof(float), hipMemcpyDeviceToDevice, stream);
}
